// GRU_326417514599
// MI455X (gfx1250) — matrix operand order repaired and checked by reading, not compiled, not run on hardware
//
#include <hip/hip_runtime.h>
#include <hip/hip_bf16.h>

typedef _Float16 h16;
typedef __attribute__((ext_vector_type(16))) _Float16 v16h;
typedef __attribute__((ext_vector_type(8)))  _Float16 v8h;
typedef __attribute__((ext_vector_type(4)))  _Float16 v4h;
typedef __attribute__((ext_vector_type(8)))  float    v8f;
typedef __attribute__((ext_vector_type(4)))  float    v4f;

#define HD    64            // hidden dim
#define IDIM  4             // input dim
#define LNUM  4             // layers
#define ODIM  5             // output dim
#define BSZ   256           // batch
#define TSZ   2048          // seq len
#define G3    192           // 3*H
#define BS    16            // batch rows per block (== WMMA M)
#define NTHR  384           // 12 waves: one N-tile (16 cols) per wave

__device__ __forceinline__ v16h load_afrag(const h16* row, int base0) {
    // 16 halves: cols [base0..base0+7] and [base0+16..base0+23]
    v8h lo = *(const v8h*)(row + base0);
    v8h hi = *(const v8h*)(row + base0 + 16);
    return __builtin_shufflevector(lo, hi, 0, 1, 2, 3, 4, 5, 6, 7,
                                           8, 9, 10, 11, 12, 13, 14, 15);
}

__launch_bounds__(NTHR)
__global__ void gru_persistent(const float* __restrict__ x,
                               const float* __restrict__ w_ih0,
                               const float* __restrict__ w_ih_rest,
                               const float* __restrict__ w_hh,
                               const float* __restrict__ b_ih,
                               const float* __restrict__ b_hh,
                               const float* __restrict__ fc_w,
                               const float* __restrict__ fc_b,
                               float* __restrict__ out,
                               h16* __restrict__ seq)
{
    // Weights transposed into B-matrix-friendly [K][N] layout, f16.
    __shared__ h16   s_wih[HD][G3];      // 24 KB
    __shared__ h16   s_whh[HD][G3];      // 24 KB
    __shared__ h16   s_x16[BS][HD];      // layer input at time t (f16)
    __shared__ h16   s_h16[BS][HD];      // hidden state (f16 feed for WMMA A)
    __shared__ float s_gx [BS][G3];      // x-side gate pre-activations
    __shared__ float s_gh [BS][G3];      // h-side gate pre-activations

    const int tid  = threadIdx.x;
    const int wave = tid >> 5;           // 0..11 -> N tile index
    const int lane = tid & 31;
    const int m    = lane & 15;          // M row (batch row in slice) / N col
    const int hi   = lane >> 4;          // lane-group (0/1)
    const int b0   = blockIdx.x * BS;

    // gate-thread mapping: tid<256 owns 4 contiguous hidden channels
    const bool is_gate = (tid < BS * HD / 4);        // 256 threads
    const int  gb  = (tid * 4) >> 6;                 // batch row in slice
    const int  gj  = (tid * 4) & 63;                 // first channel
    // loader-thread mapping: tid in [256,384) prefetches x_{t+1}
    const int  tl  = tid - 256;                      // 0..127
    const int  lb  = tl >> 3;                        // batch row
    const int  lj  = (tl & 7) * 8;                   // first of 8 halves

    // zero x staging once (layer-0 only writes cols 0..3 of each row)
    for (int idx = tid; idx < BS * HD; idx += NTHR) {
        s_x16[idx / HD][idx % HD] = (h16)0.f;
    }

    float hreg[4] = {0.f, 0.f, 0.f, 0.f};            // f32 hidden (gate threads)

    for (int l = 0; l < LNUM; ++l) {
        const bool last_layer = (l == LNUM - 1);

        // ---- stage this layer's weights into LDS: s_w[k][g] = W[g][k] ----
        for (int idx = tid; idx < HD * G3; idx += NTHR) {
            const int k = idx / G3, g = idx % G3;
            float wi;
            if (l == 0) wi = (k < IDIM) ? w_ih0[g * IDIM + k] : 0.f;
            else        wi = w_ih_rest[((size_t)(l - 1) * G3 + g) * HD + k];
            s_wih[k][g] = (h16)wi;
            s_whh[k][g] = (h16)w_hh[((size_t)l * G3 + g) * HD + k];
        }
        __syncthreads();

        // ---- hoist constant B fragments (w_hh, w_ih) into registers ----
        // B 32x16 f16 layout: lane holds N = lane%16; K = 32*chunk + 8*hi + i (+8 for i>=8)
        const int ncol = wave * 16 + m;
        v16h bW[2], bI[2];
        #pragma unroll
        for (int c = 0; c < 2; ++c) {
            #pragma unroll
            for (int i = 0; i < 16; ++i) {
                const int k = 32 * c + 8 * hi + i + (i >= 8 ? 8 : 0);
                bW[c][i] = s_whh[k][ncol];
                bI[c][i] = s_wih[k][ncol];
            }
        }

        // ---- hoist biases + reset hidden state (gate threads) ----
        float bir[4], biz[4], bin_[4], bhr[4], bhz[4], bhn[4];
        if (is_gate) {
            #pragma unroll
            for (int i = 0; i < 4; ++i) {
                bir[i]  = b_ih[l * G3 + gj + i];
                biz[i]  = b_ih[l * G3 + HD + gj + i];
                bin_[i] = b_ih[l * G3 + 2 * HD + gj + i];
                bhr[i]  = b_hh[l * G3 + gj + i];
                bhz[i]  = b_hh[l * G3 + HD + gj + i];
                bhn[i]  = b_hh[l * G3 + 2 * HD + gj + i];
                hreg[i] = 0.f;
            }
            v4h z4 = {};
            *(v4h*)&s_h16[gb][gj] = z4;
        }
        // ---- preload x_0 for this layer ----
        if (l == 0) {
            if (tl >= 0 && tl < BS * IDIM) {
                const int b = tl / IDIM, c = tl % IDIM;
                s_x16[b][c] = (h16)x[((size_t)(b0 + b) * TSZ + 0) * IDIM + c];
            }
        } else {
            if (tl >= 0) {
                *(v8h*)&s_x16[lb][lj] =
                    *(const v8h*)&seq[((size_t)(b0 + lb) * TSZ + 0) * HD + lj];
            }
        }

        // ---- sequential scan over time ----
        for (int t = 0; t < TSZ; ++t) {
            __syncthreads();   // x_t staged, h writes from t-1 visible,
                               // gate reads of s_gx/s_gh from t-1 done

            // A fragments (16x32 f16): issue all LDS loads, then 4 WMMAs
            const int baseA = 8 * hi;
            v16h aH0 = load_afrag(&s_h16[m][0], baseA);
            v16h aH1 = load_afrag(&s_h16[m][0], baseA + 32);
            v16h aX0 = load_afrag(&s_x16[m][0], baseA);
            v16h aX1 = load_afrag(&s_x16[m][0], baseA + 32);
            v8f accH = {}, accX = {};
            accH = __builtin_amdgcn_wmma_f32_16x16x32_f16(
                       false, aH0, false, bW[0], (short)0, accH, false, false);
            accX = __builtin_amdgcn_wmma_f32_16x16x32_f16(
                       false, aX0, false, bI[0], (short)0, accX, false, false);
            accH = __builtin_amdgcn_wmma_f32_16x16x32_f16(
                       false, aH1, false, bW[1], (short)0, accH, false, false);
            accX = __builtin_amdgcn_wmma_f32_16x16x32_f16(
                       false, aX1, false, bI[1], (short)0, accX, false, false);

            // scatter accumulators: VGPR v, lane -> M = v + 8*hi, N = ncol
            #pragma unroll
            for (int v = 0; v < 8; ++v) {
                const int row = v + 8 * hi;
                s_gx[row][ncol] = accX[v];
                s_gh[row][ncol] = accH[v];
            }
            __syncthreads();

            // -- waves 0-7: gate math + state update (vectorized, f32) --
            if (is_gate) {
                v4f gxr = *(const v4f*)&s_gx[gb][gj];
                v4f gxz = *(const v4f*)&s_gx[gb][gj + HD];
                v4f gxn = *(const v4f*)&s_gx[gb][gj + 2 * HD];
                v4f ghr = *(const v4f*)&s_gh[gb][gj];
                v4f ghz = *(const v4f*)&s_gh[gb][gj + HD];
                v4f ghn = *(const v4f*)&s_gh[gb][gj + 2 * HD];
                v4h hv;
                #pragma unroll
                for (int i = 0; i < 4; ++i) {
                    const float r = 1.f / (1.f + __expf(-(gxr[i] + bir[i] +
                                                          ghr[i] + bhr[i])));
                    const float z = 1.f / (1.f + __expf(-(gxz[i] + biz[i] +
                                                          ghz[i] + bhz[i])));
                    const float n = tanhf(gxn[i] + bin_[i] +
                                          r * (ghn[i] + bhn[i]));
                    hreg[i] = (1.f - z) * n + z * hreg[i];
                    hv[i] = (h16)hreg[i];
                }
                *(v4h*)&s_h16[gb][gj] = hv;
                if (!last_layer)   // in-place: next layer reads this at time t
                    *(v4h*)&seq[((size_t)(b0 + gb) * TSZ + t) * HD + gj] = hv;
            } else {
                // -- waves 8-11: prefetch x_{t+1} into LDS staging --
                if (t + 1 < TSZ) {
                    if (l == 0) {
                        if (tl < BS * IDIM) {
                            const int b = tl / IDIM, c = tl % IDIM;
                            s_x16[b][c] = (h16)
                                x[((size_t)(b0 + b) * TSZ + (t + 1)) * IDIM + c];
                        }
                    } else {
                        *(v8h*)&s_x16[lb][lj] = *(const v8h*)
                            &seq[((size_t)(b0 + lb) * TSZ + (t + 1)) * HD + lj];
                    }
                }
            }
        }
        __syncthreads();
    }

    // ---- FC head on final hidden state of last layer ----
    // dump f32 hidden registers into s_gx (reused as [BS][>=HD] buffer)
    if (is_gate) {
        #pragma unroll
        for (int i = 0; i < 4; ++i) s_gx[gb][gj + i] = hreg[i];
    }
    __syncthreads();
    if (tid < BS * ODIM) {
        const int b = tid / ODIM, o = tid % ODIM;
        float acc = fc_b[o];
        #pragma unroll
        for (int j = 0; j < HD; ++j)
            acc += s_gx[b][j] * fc_w[o * HD + j];
        out[(size_t)(b0 + b) * ODIM + o] = acc;
    }
}

extern "C" void kernel_launch(void* const* d_in, const int* in_sizes, int n_in,
                              void* d_out, int out_size, void* d_ws, size_t ws_size,
                              hipStream_t stream) {
    (void)in_sizes; (void)n_in; (void)out_size; (void)ws_size;
    const float* x        = (const float*)d_in[0];
    const float* w_ih0    = (const float*)d_in[1];
    const float* w_ih_r   = (const float*)d_in[2];
    const float* w_hh     = (const float*)d_in[3];
    const float* b_ih     = (const float*)d_in[4];
    const float* b_hh     = (const float*)d_in[5];
    const float* fc_w     = (const float*)d_in[6];
    const float* fc_b     = (const float*)d_in[7];
    float*       out      = (float*)d_out;
    h16*         seq      = (h16*)d_ws;   // needs B*T*H*2 = 64 MB

    gru_persistent<<<BSZ / BS, NTHR, 0, stream>>>(
        x, w_ih0, w_ih_r, w_hh, b_ih, b_hh, fc_w, fc_b, out, seq);
}
